// _Attention_19542101197184
// MI455X (gfx1250) — matrix operand order repaired and checked by reading, not compiled, not run on hardware
//
#include <hip/hip_runtime.h>
#include <hip/hip_bf16.h>

#define BATCH 32
#define SEQ 4096
#define DH 512          // DEC_HID == ENC_FEAT
#define AWC 1024        // attn_w row length (ENC_FEAT + DEC_HID)

typedef __attribute__((ext_vector_type(16))) __bf16 v16bf;
typedef __attribute__((ext_vector_type(8)))  float  v8f;

__device__ __forceinline__ float fast_tanh(float x) {
#if __has_builtin(__builtin_amdgcn_tanhf)
    return __builtin_amdgcn_tanhf(x);      // V_TANH_F32 (gfx1250 trans op)
#else
    return tanhf(x);
#endif
}

// ---------------------------------------------------------------------------
// Kernel 1: h_proj[b,d] = hidden[b,:] . attn_w[d, 0:512] + attn_b[d]
// ---------------------------------------------------------------------------
__global__ __launch_bounds__(256) void hproj_kernel(
    const float* __restrict__ hidden,   // (32, 512)
    const float* __restrict__ attn_w,   // (512, 1024)
    const float* __restrict__ attn_b,   // (512,)
    float* __restrict__ h_proj)         // (32, 512)
{
    int idx = blockIdx.x * 256 + threadIdx.x;   // 0 .. 16383
    int b = idx >> 9;
    int d = idx & (DH - 1);
    const float4* hrow = (const float4*)(hidden + b * DH);
    const float4* wrow = (const float4*)(attn_w + (size_t)d * AWC);  // w_hid row
    float acc = 0.f;
#pragma unroll 8
    for (int e = 0; e < DH / 4; ++e) {
        float4 h = hrow[e], w = wrow[e];
        acc += h.x * w.x + h.y * w.y + h.z * w.z + h.w * w.w;
    }
    h_proj[idx] = acc + attn_b[d];
}

// ---------------------------------------------------------------------------
// Kernel 2: scores[b,s] = v . tanh( w_enc @ enc[b,s,:] + h_proj[b,:] )
// Block: 256 threads = 8 waves covering (batch b, 64 seq rows, 512 D).
// Wave (msub, dq): msub = 32-seq-row subtile, dq = 128-wide D quarter.
// Each wave holds TWO 16-row A tiles so every B fragment feeds 2 WMMAs
// (halves w_enc L2 traffic + cvt work per FLOP).
// ---------------------------------------------------------------------------
__global__ __launch_bounds__(256) void scores_kernel(
    const float* __restrict__ enc,      // (SEQ, BATCH, DH) : s-major
    const float* __restrict__ attn_w,   // (512, 1024), w_enc at col offset 512
    const float* __restrict__ h_proj,   // (32, 512)
    const float* __restrict__ v_w,      // (512,)
    float* __restrict__ scores)         // (32, SEQ)
{
    __shared__ float lds_part[8 * 32];  // [wave][local m 0..31]

    const int blk    = blockIdx.x;      // 2048 blocks
    const int b      = blk & (BATCH - 1);
    const int schunk = blk >> 5;        // 0..63, 64 seq rows each
    const int wave   = threadIdx.x >> 5;
    const int lane   = threadIdx.x & 31;
    const int msub   = wave >> 2;       // 0/1 : which 32-row seq subtile
    const int dq     = wave & 3;        // 0..3 : 128-wide D quarter
    const int lane16 = lane & 15;
    const bool lo    = lane < 16;

    const int s0 = schunk * 64 + msub * 32;
    const float* encrow0 = enc + (size_t)(s0 + lane16) * (BATCH * DH) + (size_t)b * DH;
    const float* encrow1 = encrow0 + (size_t)16 * (BATCH * DH);
    // B: lane holds N = lane16 -> w_enc row d = dq*128 + dt*16 + lane16
    const float* wbase = attn_w + (size_t)(dq * 128 + lane16) * AWC + DH;

    v8f acc0[8], acc1[8];
#pragma unroll
    for (int dt = 0; dt < 8; ++dt) { acc0[dt] = (v8f)(0.f); acc1[dt] = (v8f)(0.f); }

    for (int kt = 0; kt < 16; ++kt) {               // K = 512 in steps of 32
        // ---- A tiles: 16x32 bf16 (M=seq, K), ISA lane layout ----
        const int kbA = kt * 32 + (lo ? 0 : 8);
        float fa0[16], fa1[16];
        {
            const float4* p0 = (const float4*)(encrow0 + kbA);
            const float4* q0 = (const float4*)(encrow0 + kbA + 16);
            *(float4*)(fa0 + 0)  = p0[0];  *(float4*)(fa0 + 4)  = p0[1];
            *(float4*)(fa0 + 8)  = q0[0];  *(float4*)(fa0 + 12) = q0[1];
            const float4* p1 = (const float4*)(encrow1 + kbA);
            const float4* q1 = (const float4*)(encrow1 + kbA + 16);
            *(float4*)(fa1 + 0)  = p1[0];  *(float4*)(fa1 + 4)  = p1[1];
            *(float4*)(fa1 + 8)  = q1[0];  *(float4*)(fa1 + 12) = q1[1];
        }
        v16bf av0, av1;
#pragma unroll
        for (int i = 0; i < 16; ++i) { av0[i] = (__bf16)fa0[i]; av1[i] = (__bf16)fa1[i]; }

        const int kbB = kt * 32 + (lo ? 0 : 8);
#pragma unroll
        for (int dt = 0; dt < 8; ++dt) {
            // ---- B tile: 32x16 bf16 (K x N), N = lane16, K in ISA table order ----
            float fb[16];
            const float4* pb = (const float4*)(wbase + (size_t)dt * 16 * AWC + kbB);
            *(float4*)(fb + 0)  = pb[0];
            *(float4*)(fb + 4)  = pb[1];
            *(float4*)(fb + 8)  = pb[4];
            *(float4*)(fb + 12) = pb[5];
            v16bf bv;
#pragma unroll
            for (int i = 0; i < 16; ++i) bv[i] = (__bf16)fb[i];

            acc0[dt] = __builtin_amdgcn_wmma_f32_16x16x32_bf16(
                false, av0, false, bv, (short)0, acc0[dt], false, false);
            acc1[dt] = __builtin_amdgcn_wmma_f32_16x16x32_bf16(
                false, av1, false, bv, (short)0, acc1[dt], false, false);
        }
    }

    // ---- epilogue: tanh(acc + h_proj) * v, reduce over D ----
    // C/D layout: lane -> N = lane16, VGPR r -> M = r + (lo ? 0 : 8)
    float part0[8], part1[8];
#pragma unroll
    for (int r = 0; r < 8; ++r) { part0[r] = 0.f; part1[r] = 0.f; }

#pragma unroll
    for (int dt = 0; dt < 8; ++dt) {
        const int d  = dq * 128 + dt * 16 + lane16;
        const float hp = h_proj[b * DH + d];
        const float vv = v_w[d];
#pragma unroll
        for (int r = 0; r < 8; ++r) {
            part0[r] += fast_tanh(acc0[dt][r] + hp) * vv;
            part1[r] += fast_tanh(acc1[dt][r] + hp) * vv;
        }
    }

    // reduce over the 16 lanes sharing the same M (xor stays inside the half)
#pragma unroll
    for (int r = 0; r < 8; ++r) {
        float v0 = part0[r], v1 = part1[r];
        v0 += __shfl_xor(v0, 1, 32);  v1 += __shfl_xor(v1, 1, 32);
        v0 += __shfl_xor(v0, 2, 32);  v1 += __shfl_xor(v1, 2, 32);
        v0 += __shfl_xor(v0, 4, 32);  v1 += __shfl_xor(v1, 4, 32);
        v0 += __shfl_xor(v0, 8, 32);  v1 += __shfl_xor(v1, 8, 32);
        part0[r] = v0; part1[r] = v1;
    }
    if (lane16 == 0) {
        const int mhalf = lo ? 0 : 8;
#pragma unroll
        for (int r = 0; r < 8; ++r) {
            lds_part[wave * 32 + mhalf + r]      = part0[r];
            lds_part[wave * 32 + 16 + mhalf + r] = part1[r];
        }
    }
    __syncthreads();

    // deterministic cross-wave (D-quarter) reduction, then store 64 scores
    if (threadIdx.x < 64) {
        const int msel = threadIdx.x >> 5;      // which 32-row subtile
        const int mm   = threadIdx.x & 31;
        float sum = 0.f;
#pragma unroll
        for (int q = 0; q < 4; ++q)
            sum += lds_part[(msel * 4 + q) * 32 + mm];
        scores[(size_t)b * SEQ + schunk * 64 + threadIdx.x] = sum;
    }
}

// ---------------------------------------------------------------------------
// Kernel 3: row softmax over 4096 elements, one block per batch row
// ---------------------------------------------------------------------------
__global__ __launch_bounds__(256) void softmax_kernel(
    const float* __restrict__ scores, float* __restrict__ out)
{
    __shared__ float red_max[8];
    __shared__ float red_sum[8];
    const int b    = blockIdx.x;
    const int tid  = threadIdx.x;
    const int lane = tid & 31;
    const int wv   = tid >> 5;
    const float* row = scores + (size_t)b * SEQ;

    float vals[16];
    float mx = -3.402823466e38f;
#pragma unroll
    for (int i = 0; i < 16; ++i) {
        vals[i] = row[tid + i * 256];
        mx = fmaxf(mx, vals[i]);
    }
#pragma unroll
    for (int off = 1; off < 32; off <<= 1)
        mx = fmaxf(mx, __shfl_xor(mx, off, 32));
    if (lane == 0) red_max[wv] = mx;
    __syncthreads();
    float bmax = red_max[0];
#pragma unroll
    for (int i = 1; i < 8; ++i) bmax = fmaxf(bmax, red_max[i]);

    float s = 0.f;
#pragma unroll
    for (int i = 0; i < 16; ++i) {
        vals[i] = __expf(vals[i] - bmax);
        s += vals[i];
    }
#pragma unroll
    for (int off = 1; off < 32; off <<= 1)
        s += __shfl_xor(s, off, 32);
    if (lane == 0) red_sum[wv] = s;
    __syncthreads();
    float tot = 0.f;
#pragma unroll
    for (int i = 0; i < 8; ++i) tot += red_sum[i];
    const float inv = 1.f / tot;
#pragma unroll
    for (int i = 0; i < 16; ++i)
        out[(size_t)b * SEQ + tid + i * 256] = vals[i] * inv;
}

// ---------------------------------------------------------------------------
extern "C" void kernel_launch(void* const* d_in, const int* in_sizes, int n_in,
                              void* d_out, int out_size, void* d_ws, size_t ws_size,
                              hipStream_t stream) {
    const float* hidden = (const float*)d_in[0];   // (32, 512)
    const float* enc    = (const float*)d_in[1];   // (4096, 32, 512)
    const float* attn_w = (const float*)d_in[2];   // (512, 1024)
    const float* attn_b = (const float*)d_in[3];   // (512,)
    const float* v_w    = (const float*)d_in[4];   // (1, 512)
    float* out = (float*)d_out;                    // (32, 4096)

    float* h_proj = (float*)d_ws;                  // 32*512 floats (64 KB)
    float* scores = h_proj + BATCH * DH;           // 32*4096 floats (512 KB)

    hproj_kernel<<<(BATCH * DH) / 256, 256, 0, stream>>>(hidden, attn_w, attn_b, h_proj);
    scores_kernel<<<BATCH * (SEQ / 64), 256, 0, stream>>>(enc, attn_w, h_proj, v_w, scores);
    softmax_kernel<<<BATCH, 256, 0, stream>>>(scores, out);
}
